// MoELayer_91147795955940
// MI455X (gfx1250) — compile-verified, matrix operand order checked by reading, not run on hardware
//
#include <hip/hip_runtime.h>
#include <hip/hip_bf16.h>
#include <stdint.h>

#define H_DIM 1024
#define I_DIM 1024
#define NEXP  16
#define TOPK  4
#define T_TOK 2048
#define SH_OFF (TOPK * T_TOK)   // act row offset for shared expert

#define KP    32                // K panel (one WMMA k-step)
#define P_STR 40                // panel row stride in bf16 elements (32 + 8 pad)

typedef __attribute__((ext_vector_type(16))) __bf16 v16bf;
typedef __attribute__((ext_vector_type(8)))  float  v8f;
typedef __attribute__((ext_vector_type(4)))  int    v4i;

union Frag  { uint32_t u[8]; v16bf v; };
union Pack2 { __bf16 h[2]; uint32_t u; };

#define GAS __attribute__((address_space(1)))
#define LAS __attribute__((address_space(3)))

#if defined(__HIP_DEVICE_COMPILE__) && \
    __has_builtin(__builtin_amdgcn_global_load_async_to_lds_b128) && \
    __has_builtin(__builtin_amdgcn_s_wait_asynccnt)
#define HAVE_ASYNC 1
#else
#define HAVE_ASYNC 0
#endif

#if HAVE_ASYNC
__device__ inline void async_b128(const void* g, void* l) {
  // prototype (from clang diag): (v4i addrspace(1)*, v4i addrspace(3)*, imm offset, imm cpol)
  __builtin_amdgcn_global_load_async_to_lds_b128(
      (GAS v4i*)(GAS void*)g, (LAS v4i*)(LAS void*)l, 0, 0);
}
#endif

// 16x32 bf16 fragment from an LDS tile (A layout; also valid for B taken from
// an [N][K] row-major tile). Lane L: row = row0 + L%16, half g = L/16; dwords
// at element offsets {0,2,4,6,16,18,20,22} + 8g  ->  two ds_load_b128 per frag.
__device__ inline v16bf load_frag16(const __bf16* base, int row0, int stride, int k0) {
  int lane = threadIdx.x & 31;
  int r = row0 + (lane & 15);
  int g = lane >> 4;
  const uint32_t* p = (const uint32_t*)(base + (size_t)r * stride + k0 + g * 8);
  Frag f;
  f.u[0] = p[0];  f.u[1] = p[1];  f.u[2] = p[2];  f.u[3] = p[3];
  f.u[4] = p[8];  f.u[5] = p[9];  f.u[6] = p[10]; f.u[7] = p[11];
  return f.v;
}

__device__ inline v8f wmma_bf16(v16bf a, v16bf b, v8f c) {
  return __builtin_amdgcn_wmma_f32_16x16x32_bf16(false, a, false, b, (short)0, c, false, false);
}

__device__ inline float silu_mul(float t1, float t3, float w) {
  float e = __expf(-t1);                       // v_exp_f32
  return t1 * __builtin_amdgcn_rcpf(1.0f + e) * t3 * w;  // v_rcp_f32, no IEEE div
}

// ---------------- init: zero output + counters ----------------
__global__ __launch_bounds__(256) void moe_init(float* __restrict__ out,
                                                int* __restrict__ counts, int n) {
  int i = blockIdx.x * 256 + threadIdx.x;
  if (i < n) out[i] = 0.0f;
  if (i < NEXP) counts[i] = 0;
}

// ---------------- router ----------------
__global__ __launch_bounds__(256) void moe_router(
    const float* __restrict__ x, const float* __restrict__ gw,
    const float* __restrict__ bias, int* __restrict__ counts,
    int* __restrict__ l_tok, int* __restrict__ l_row, float* __restrict__ l_w) {
  int t = blockIdx.x * 256 + threadIdx.x;
  if (t >= T_TOK) return;
  float logit[NEXP];
#pragma unroll
  for (int e = 0; e < NEXP; ++e) logit[e] = 0.0f;
  const float* xr = x + (size_t)t * H_DIM;
  for (int h = 0; h < H_DIM; ++h) {
    float xv = xr[h];
#pragma unroll
    for (int e = 0; e < NEXP; ++e) logit[e] += xv * gw[e * H_DIM + h];  // uniform -> s_load
  }
  float biased[NEXP];
#pragma unroll
  for (int e = 0; e < NEXP; ++e) biased[e] = logit[e] + bias[e];

  int sel[TOPK]; float raw[TOPK];
#pragma unroll
  for (int k = 0; k < TOPK; ++k) {
    int best = 0; float bv = -3.4e38f;
#pragma unroll
    for (int e = 0; e < NEXP; ++e)
      if (biased[e] > bv) { bv = biased[e]; best = e; }   // first index wins ties
    sel[k] = best; raw[k] = logit[best];
    biased[best] = -3.4e38f;
  }
  float mx = raw[0];
#pragma unroll
  for (int k = 1; k < TOPK; ++k) mx = fmaxf(mx, raw[k]);
  float wk[TOPK], s = 0.0f;
#pragma unroll
  for (int k = 0; k < TOPK; ++k) { wk[k] = __expf(raw[k] - mx); s += wk[k]; }
  float inv = __builtin_amdgcn_rcpf(s);
#pragma unroll
  for (int k = 0; k < TOPK; ++k) {
    int e = sel[k];
    int slot = atomicAdd(&counts[e], 1);
    l_tok[e * T_TOK + slot] = t;
    l_row[e * T_TOK + slot] = t * TOPK + k;
    l_w  [e * T_TOK + slot] = wk[k] * inv;
  }
}

// ---------------- cast x -> bf16 ----------------
__global__ __launch_bounds__(256) void moe_cast(const float* __restrict__ x,
                                                __bf16* __restrict__ xh, int n4) {
  int i = blockIdx.x * 256 + threadIdx.x;
  if (i >= n4) return;
  float4 v = ((const float4*)x)[i];
  Pack2 a, b;
  a.h[0] = (__bf16)v.x; a.h[1] = (__bf16)v.y;
  b.h[0] = (__bf16)v.z; b.h[1] = (__bf16)v.w;
  ((uint32_t*)xh)[i * 2]     = a.u;
  ((uint32_t*)xh)[i * 2 + 1] = b.u;
}

// ---------------- grouped GEMM1: act = silu(x@w1^T) * (x@w3^T) * route_w ----
// grid: (NEXP+1 groups, 64 token tiles, 2 N-halves), block 256 (8 waves).
// Block tile: M=32, N-chunk=256 (wave owns N=32), K panel=32.
__global__ __launch_bounds__(256) void moe_gemm1(
    const __bf16* __restrict__ xh,
    const float* __restrict__ w1, const float* __restrict__ w3,
    const float* __restrict__ sw1, const float* __restrict__ sw3,
    const int* __restrict__ counts, const int* __restrict__ l_tok,
    const int* __restrict__ l_row, const float* __restrict__ l_w,
    __bf16* __restrict__ act) {
  const int g  = blockIdx.x;
  const int mt = blockIdx.y;
  const int zn = blockIdx.z;
  int valid;
  const float *W1, *W3;
  if (g == NEXP) { W1 = sw1; W3 = sw3; valid = 32; }
  else {
    int c = counts[g];
    if (mt * 32 >= c) return;                 // uniform early exit
    valid = c - mt * 32; if (valid > 32) valid = 32;
    W1 = w1 + (size_t)g * I_DIM * H_DIM;
    W3 = w3 + (size_t)g * I_DIM * H_DIM;
  }

  __shared__ __align__(16) __bf16 Xp[32 * P_STR];        // 2.5 KB
  __shared__ __align__(16) __bf16 Wp0[256 * P_STR];      // 20 KB
  __shared__ __align__(16) __bf16 Wp1[256 * P_STR];      // 20 KB
  __shared__ int   s_tok[32], s_arow[32];
  __shared__ float s_w[32];

  const int tid = threadIdx.x;
  if (tid < 32) {
    int tok, arow; float wt;
    if (g == NEXP) { tok = mt * 32 + tid; arow = SH_OFF + tok; wt = 1.0f; }
    else if (tid < valid) {
      int s = g * T_TOK + mt * 32 + tid;
      tok = l_tok[s]; arow = l_row[s]; wt = l_w[s];
    } else { tok = 0; arow = 0; wt = 0.0f; }
    s_tok[tid] = tok; s_arow[tid] = arow; s_w[tid] = wt;
  }

  const int nl   = (tid >> 5) * 32;   // wave's N base within 256-chunk
  const int lane = tid & 31;

  for (int nc = zn * 2; nc < zn * 2 + 2; ++nc) {
    v8f acc1[2][2] = {}, acc3[2][2] = {};
    const int n0 = nc * 256;
    for (int kp = 0; kp < H_DIM / KP; ++kp) {
      const int k0 = kp * KP;
      __syncthreads();
      // stage X panel (gathered bf16 rows): 32 rows x 64 B
#if HAVE_ASYNC
      if (tid < 128) {
        int r = tid >> 2, c = tid & 3;   // 4 x 16 B chunks per row
        async_b128(xh + (size_t)s_tok[r] * H_DIM + k0 + c * 8,
                   Xp + r * P_STR + c * 8);
      }
#else
      for (int idx = tid; idx < 32 * 16; idx += 256) {
        int r = idx >> 4, p2 = idx & 15;
        ((uint32_t*)(Xp + r * P_STR))[p2] =
            *(const uint32_t*)(xh + (size_t)s_tok[r] * H_DIM + k0 + p2 * 2);
      }
#endif
      // stage w1/w3 panels fp32 -> bf16 (float4 loads, 8 iters/thread)
      for (int idx = tid; idx < 256 * 8; idx += 256) {
        int n = idx >> 3, p4 = idx & 7;
        size_t go = (size_t)(n0 + n) * H_DIM + k0 + p4 * 4;
        float4 a = *(const float4*)(W1 + go);
        float4 b = *(const float4*)(W3 + go);
        Pack2 a0, a1, b0, b1;
        a0.h[0] = (__bf16)a.x; a0.h[1] = (__bf16)a.y;
        a1.h[0] = (__bf16)a.z; a1.h[1] = (__bf16)a.w;
        b0.h[0] = (__bf16)b.x; b0.h[1] = (__bf16)b.y;
        b1.h[0] = (__bf16)b.z; b1.h[1] = (__bf16)b.w;
        uint32_t* d1 = (uint32_t*)(Wp0 + n * P_STR + p4 * 4);
        d1[0] = a0.u; d1[1] = a1.u;
        uint32_t* d3 = (uint32_t*)(Wp1 + n * P_STR + p4 * 4);
        d3[0] = b0.u; d3[1] = b1.u;
      }
#if HAVE_ASYNC
      __builtin_amdgcn_s_wait_asynccnt(0);
#endif
      __syncthreads();

      v16bf a0 = load_frag16(Xp, 0,  P_STR, 0);
      v16bf a1 = load_frag16(Xp, 16, P_STR, 0);
      v16bf b1lo = load_frag16(Wp0, nl,      P_STR, 0);
      v16bf b1hi = load_frag16(Wp0, nl + 16, P_STR, 0);
      v16bf b3lo = load_frag16(Wp1, nl,      P_STR, 0);
      v16bf b3hi = load_frag16(Wp1, nl + 16, P_STR, 0);
      acc1[0][0] = wmma_bf16(a0, b1lo, acc1[0][0]);
      acc1[0][1] = wmma_bf16(a0, b1hi, acc1[0][1]);
      acc1[1][0] = wmma_bf16(a1, b1lo, acc1[1][0]);
      acc1[1][1] = wmma_bf16(a1, b1hi, acc1[1][1]);
      acc3[0][0] = wmma_bf16(a0, b3lo, acc3[0][0]);
      acc3[0][1] = wmma_bf16(a0, b3hi, acc3[0][1]);
      acc3[1][0] = wmma_bf16(a1, b3lo, acc3[1][0]);
      acc3[1][1] = wmma_bf16(a1, b3hi, acc3[1][1]);
    }
    // epilogue: silu(t1)*t3*route_w -> bf16 act rows
    const int lane15 = lane & 15;
    const int mhi = (lane >> 4) * 8;
#pragma unroll
    for (int tm = 0; tm < 2; ++tm) {
#pragma unroll
      for (int tn = 0; tn < 2; ++tn) {
        int ng = n0 + nl + tn * 16 + lane15;
#pragma unroll
        for (int r = 0; r < 8; ++r) {
          int m = tm * 16 + r + mhi;
          if (m < valid) {
            float av = silu_mul(acc1[tm][tn][r], acc3[tm][tn][r], s_w[m]);
            act[(size_t)s_arow[m] * I_DIM + ng] = (__bf16)av;
          }
        }
      }
    }
  }
}

// ---------------- grouped GEMM2: out += act @ w2^T ----------------
__global__ __launch_bounds__(256) void moe_gemm2(
    const __bf16* __restrict__ act,
    const float* __restrict__ w2, const float* __restrict__ sw2,
    const int* __restrict__ counts, const int* __restrict__ l_tok,
    const int* __restrict__ l_row, float* __restrict__ out) {
  const int g  = blockIdx.x;
  const int mt = blockIdx.y;
  const int zn = blockIdx.z;
  int valid;
  const float* W2;
  if (g == NEXP) { W2 = sw2; valid = 32; }
  else {
    int c = counts[g];
    if (mt * 32 >= c) return;
    valid = c - mt * 32; if (valid > 32) valid = 32;
    W2 = w2 + (size_t)g * H_DIM * I_DIM;
  }

  __shared__ __align__(16) __bf16 Ap[32 * P_STR];
  __shared__ __align__(16) __bf16 Wp[256 * P_STR];
  __shared__ int s_tok[32], s_arow[32];

  const int tid = threadIdx.x;
  if (tid < 32) {
    int tok, arow;
    if (g == NEXP) { tok = mt * 32 + tid; arow = SH_OFF + tok; }
    else if (tid < valid) {
      int s = g * T_TOK + mt * 32 + tid;
      tok = l_tok[s]; arow = l_row[s];
    } else { tok = 0; arow = 0; }
    s_tok[tid] = tok; s_arow[tid] = arow;
  }

  const int nl   = (tid >> 5) * 32;
  const int lane = tid & 31;

  for (int nc = zn * 2; nc < zn * 2 + 2; ++nc) {
    v8f acc[2][2] = {};
    const int n0 = nc * 256;
    for (int kp = 0; kp < I_DIM / KP; ++kp) {
      const int k0 = kp * KP;
      __syncthreads();
#if HAVE_ASYNC
      if (tid < 128) {
        int r = tid >> 2, c = tid & 3;
        async_b128(act + (size_t)s_arow[r] * I_DIM + k0 + c * 8,
                   Ap + r * P_STR + c * 8);
      }
#else
      for (int idx = tid; idx < 32 * 16; idx += 256) {
        int r = idx >> 4, p2 = idx & 15;
        ((uint32_t*)(Ap + r * P_STR))[p2] =
            *(const uint32_t*)(act + (size_t)s_arow[r] * I_DIM + k0 + p2 * 2);
      }
#endif
      for (int idx = tid; idx < 256 * 8; idx += 256) {
        int n = idx >> 3, p4 = idx & 7;
        size_t go = (size_t)(n0 + n) * I_DIM + k0 + p4 * 4;
        float4 a = *(const float4*)(W2 + go);
        Pack2 a0, a1;
        a0.h[0] = (__bf16)a.x; a0.h[1] = (__bf16)a.y;
        a1.h[0] = (__bf16)a.z; a1.h[1] = (__bf16)a.w;
        uint32_t* d = (uint32_t*)(Wp + n * P_STR + p4 * 4);
        d[0] = a0.u; d[1] = a1.u;
      }
#if HAVE_ASYNC
      __builtin_amdgcn_s_wait_asynccnt(0);
#endif
      __syncthreads();

      v16bf a0  = load_frag16(Ap, 0,  P_STR, 0);
      v16bf a1  = load_frag16(Ap, 16, P_STR, 0);
      v16bf blo = load_frag16(Wp, nl,      P_STR, 0);
      v16bf bhi = load_frag16(Wp, nl + 16, P_STR, 0);
      acc[0][0] = wmma_bf16(a0, blo, acc[0][0]);
      acc[0][1] = wmma_bf16(a0, bhi, acc[0][1]);
      acc[1][0] = wmma_bf16(a1, blo, acc[1][0]);
      acc[1][1] = wmma_bf16(a1, bhi, acc[1][1]);
    }
    const int lane15 = lane & 15;
    const int mhi = (lane >> 4) * 8;
#pragma unroll
    for (int tm = 0; tm < 2; ++tm) {
#pragma unroll
      for (int tn = 0; tn < 2; ++tn) {
        int ng = n0 + nl + tn * 16 + lane15;
#pragma unroll
        for (int r = 0; r < 8; ++r) {
          int m = tm * 16 + r + mhi;
          if (m < valid)
            atomicAdd(&out[(size_t)s_tok[m] * H_DIM + ng], acc[tm][tn][r]);
        }
      }
    }
  }
}

extern "C" void kernel_launch(void* const* d_in, const int* in_sizes, int n_in,
                              void* d_out, int out_size, void* d_ws, size_t ws_size,
                              hipStream_t stream) {
  const float* x   = (const float*)d_in[0];
  const float* gw  = (const float*)d_in[1];
  const float* eb  = (const float*)d_in[2];
  const float* w1  = (const float*)d_in[3];
  const float* w2  = (const float*)d_in[4];
  const float* w3  = (const float*)d_in[5];
  const float* sw1 = (const float*)d_in[6];
  const float* sw2 = (const float*)d_in[7];
  const float* sw3 = (const float*)d_in[8];
  float* out = (float*)d_out;

  char* ws = (char*)d_ws;
  size_t off = 0;
  int*    counts = (int*)(ws + off);    off += 256;
  int*    l_tok  = (int*)(ws + off);    off += (size_t)NEXP * T_TOK * 4;
  int*    l_row  = (int*)(ws + off);    off += (size_t)NEXP * T_TOK * 4;
  float*  l_w    = (float*)(ws + off);  off += (size_t)NEXP * T_TOK * 4;
  __bf16* xh     = (__bf16*)(ws + off); off += (size_t)T_TOK * H_DIM * 2;
  __bf16* act    = (__bf16*)(ws + off); off += (size_t)(TOPK + 1) * T_TOK * I_DIM * 2;

  moe_init<<<(T_TOK * H_DIM + 255) / 256, 256, 0, stream>>>(out, counts, T_TOK * H_DIM);
  moe_router<<<T_TOK / 256, 256, 0, stream>>>(x, gw, eb, counts, l_tok, l_row, l_w);
  moe_cast<<<(T_TOK * H_DIM / 4 + 255) / 256, 256, 0, stream>>>(x, xh, T_TOK * H_DIM / 4);
  moe_gemm1<<<dim3(NEXP + 1, T_TOK / 32, 2), 256, 0, stream>>>(
      xh, w1, w3, sw1, sw3, counts, l_tok, l_row, l_w, act);
  moe_gemm2<<<dim3(NEXP + 1, T_TOK / 32, 2), 256, 0, stream>>>(
      act, w2, sw2, counts, l_tok, l_row, out);
}
